// ScaledDotProductAttention_58016418234403
// MI455X (gfx1250) — matrix operand order repaired and checked by reading, not compiled, not run on hardware
//
#include <hip/hip_runtime.h>
#include <hip/hip_bf16.h>

typedef __attribute__((ext_vector_type(16))) _Float16     v16h;
typedef __attribute__((ext_vector_type(8)))  float        v8f;
typedef __attribute__((ext_vector_type(4)))  float        v4f;
typedef __attribute__((ext_vector_type(4)))  unsigned int v4u;

#define B_     32
#define LQ     2048
#define LK     2048
#define DH     64
#define BK     64            // k rows staged per chunk
#define WAVES  4             // waves per block; each wave owns 16 q rows
#define QROWS  (WAVES * 16)  // 64 q rows per block
#define PPITCH 2056          // 2048 + 8 halfs padding (bank-conflict avoidance)
#define KPITCH 72            // 64 + 8 halfs padding

// dynamic LDS layout (bytes)
#define SMEM_BYTES ((WAVES * 16 * PPITCH + BK * KPITCH + DH * KPITCH) * 2 + WAVES * 16 * 4)

union HV { v16h h; v4u q[2]; };

// build a v16h WMMA fragment from two 16-byte-aligned LDS chunks -> 2x ds_load_b128
__device__ __forceinline__ v16h ld2x16B(const _Float16* p0, const _Float16* p1) {
  HV t;
  t.q[0] = *(const v4u*)p0;
  t.q[1] = *(const v4u*)p1;
  return t.h;
}

__global__ void __launch_bounds__(128)
sdpa_fwd_kernel(const float* __restrict__ Q, const float* __restrict__ K,
                const float* __restrict__ V, const float* __restrict__ M,
                float* __restrict__ Out, float* __restrict__ Attn)
{
  extern __shared__ char smem_raw[];
  _Float16* Pbuf = (_Float16*)smem_raw;                 // WAVES*16 x PPITCH  (probs, f16)
  _Float16* Klds = Pbuf + WAVES * 16 * PPITCH;          // BK x KPITCH       (K chunk, f16)
  _Float16* Vtld = Klds + BK * KPITCH;                  // DH x KPITCH       (V chunk, transposed)
  float*    Rs   = (float*)(Vtld + DH * KPITCH);        // WAVES*16 rcp(row sums)

  const int tid  = threadIdx.x;
  const int lane = tid & 31;
  const int wave = tid >> 5;
  const int half = lane >> 4;   // which 16-lane half of the wave
  const int ln   = lane & 15;

  const int b  = blockIdx.y;
  const int q0 = blockIdx.x * QROWS;

  const float* maskp = M + (size_t)b * LK;

  // ---- Q fragments (A-matrix layout for 16x16x32: lane=row ln, K split by half) ----
  const int    qrow = q0 + wave * 16 + ln;
  const float* qp   = Q + ((size_t)b * LQ + qrow) * DH;
  const int    k0   = half * 8;
  v16h aQ0, aQ1;
  #pragma unroll
  for (int i = 0; i < 8; ++i) {
    aQ0[i]     = (_Float16)qp[k0 + i];
    aQ0[8 + i] = (_Float16)qp[k0 + 16 + i];
    aQ1[i]     = (_Float16)qp[32 + k0 + i];
    aQ1[8 + i] = (_Float16)qp[32 + k0 + 16 + i];
  }

  v8f o0 = {}, o1 = {}, o2 = {}, o3 = {};   // O accumulators, d cols {0,16,32,48}+ln
  float sumv[8];
  #pragma unroll
  for (int j = 0; j < 8; ++j) sumv[j] = 0.0f;

  // cooperative staging indices: 128 threads cover 64 rows x 64 cols, 32 floats each
  const int srow = tid >> 1;
  const int sdb  = (tid & 1) * 32;

  for (int kc = 0; kc < LK; kc += BK) {
    __syncthreads();
    {
      const float* kp = K + ((size_t)b * LK + kc + srow) * DH + sdb;
      const float* vp = V + ((size_t)b * LK + kc + srow) * DH + sdb;
      #pragma unroll
      for (int i = 0; i < 32; ++i)
        Klds[srow * KPITCH + sdb + i] = (_Float16)kp[i];
      #pragma unroll
      for (int i = 0; i < 32; ++i)
        Vtld[(sdb + i) * KPITCH + srow] = (_Float16)vp[i];     // transpose V in LDS
      if (kc + BK < LK) {                                      // global_prefetch_b8
        __builtin_prefetch(kp + (size_t)BK * DH, 0, 0);
        __builtin_prefetch(vp + (size_t)BK * DH, 0, 0);
      }
    }
    __syncthreads();

    // ---- S = Q*K^T for 16 q rows x BK k cols; exp; stash P(f16) in LDS ----
    _Float16* Pw = Pbuf + (wave * 16 + half * 8) * PPITCH;
    #pragma unroll
    for (int sub = 0; sub < 4; ++sub) {
      const _Float16* kb = Klds + (sub * 16 + ln) * KPITCH + half * 8;
      v16h bk0 = ld2x16B(kb,      kb + 16);       // d 0..31 slice (per-lane half)
      v16h bk1 = ld2x16B(kb + 32, kb + 48);       // d 32..63 slice
      v8f s = {};
      s = __builtin_amdgcn_wmma_f32_16x16x32_f16(false, aQ0, false, bk0, (short)0, s, false, false);
      s = __builtin_amdgcn_wmma_f32_16x16x32_f16(false, aQ1, false, bk1, (short)0, s, false, false);
      const int   col  = kc + sub * 16 + ln;      // C layout: lane -> one k column
      const float madd = -1e30f * (1.0f - maskp[col]);
      #pragma unroll
      for (int j = 0; j < 8; ++j) {
        float p = __expf(s[j] * 0.125f + madd);   // masked -> exp(-1e30) == 0
        sumv[j] += p;
        Pw[j * PPITCH + col] = (_Float16)p;
      }
    }

    // ---- O += P_chunk @ V_chunk (same-wave LDS ops are in-order: no barrier) ----
    const _Float16* Pa = Pbuf + (wave * 16 + ln) * PPITCH + kc + half * 16;
    #pragma unroll
    for (int h2 = 0; h2 < 2; ++h2) {
      const _Float16* pa = Pa + h2 * 32;
      v16h aP = ld2x16B(pa, pa + 8);
      const _Float16* vb = Vtld + ln * KPITCH + h2 * 32 + half * 16;
      v16h bv;
      bv = ld2x16B(vb,                   vb + 8);
      o0 = __builtin_amdgcn_wmma_f32_16x16x32_f16(false, aP, false, bv, (short)0, o0, false, false);
      bv = ld2x16B(vb + 16 * KPITCH,     vb + 16 * KPITCH + 8);
      o1 = __builtin_amdgcn_wmma_f32_16x16x32_f16(false, aP, false, bv, (short)0, o1, false, false);
      bv = ld2x16B(vb + 32 * KPITCH,     vb + 32 * KPITCH + 8);
      o2 = __builtin_amdgcn_wmma_f32_16x16x32_f16(false, aP, false, bv, (short)0, o2, false, false);
      bv = ld2x16B(vb + 48 * KPITCH,     vb + 48 * KPITCH + 8);
      o3 = __builtin_amdgcn_wmma_f32_16x16x32_f16(false, aP, false, bv, (short)0, o3, false, false);
    }
  }

  // ---- row-sum reduction across the 16 lanes of each half, then reciprocal ----
  #pragma unroll
  for (int j = 0; j < 8; ++j) {
    float t = sumv[j];
    t += __shfl_xor(t, 1, 32);
    t += __shfl_xor(t, 2, 32);
    t += __shfl_xor(t, 4, 32);
    t += __shfl_xor(t, 8, 32);
    sumv[j] = __builtin_amdgcn_rcpf(t);           // rcp(row sum) for row half*8+j
  }
  if (ln == 0) {
    #pragma unroll
    for (int j = 0; j < 8; ++j) Rs[wave * 16 + half * 8 + j] = sumv[j];
  }
  __syncthreads();

  // ---- write normalized O: row = half*8+j, col = {0,16,32,48}+ln ----
  {
    const size_t obase = ((size_t)b * LQ + q0 + wave * 16 + half * 8) * DH + ln;
    #pragma unroll
    for (int j = 0; j < 8; ++j) {
      const size_t r = obase + (size_t)j * DH;
      Out[r +  0] = o0[j] * sumv[j];
      Out[r + 16] = o1[j] * sumv[j];
      Out[r + 32] = o2[j] * sumv[j];
      Out[r + 48] = o3[j] * sumv[j];
    }
  }

  // ---- stream normalized attn (537 MB total): float4 non-temporal stores ----
  const _Float16* Pr = Pbuf + wave * 16 * PPITCH;
  #pragma unroll 1
  for (int m = 0; m < 16; ++m) {
    const float rr   = Rs[wave * 16 + m];
    float*      arow = Attn + ((size_t)b * LQ + q0 + wave * 16 + m) * (size_t)LK;
    for (int cb = lane * 4; cb < LK; cb += 128) {
      v4f val;
      #pragma unroll
      for (int i = 0; i < 4; ++i)
        val[i] = (float)Pr[m * PPITCH + cb + i] * rr;
      __builtin_nontemporal_store(val, (v4f*)(arow + cb));
    }
  }
}

extern "C" void kernel_launch(void* const* d_in, const int* in_sizes, int n_in,
                              void* d_out, int out_size, void* d_ws, size_t ws_size,
                              hipStream_t stream) {
  (void)in_sizes; (void)n_in; (void)out_size; (void)d_ws; (void)ws_size;
  const float* q    = (const float*)d_in[0];
  const float* k    = (const float*)d_in[1];
  const float* v    = (const float*)d_in[2];
  const float* mask = (const float*)d_in[3];
  float* out  = (float*)d_out;
  float* attn = out + (size_t)B_ * LQ * DH;   // tuple order: (output, attn)

  (void)hipFuncSetAttribute(reinterpret_cast<const void*>(sdpa_fwd_kernel),
                            hipFuncAttributeMaxDynamicSharedMemorySize, SMEM_BYTES);

  dim3 grid(LQ / QROWS, B_);
  dim3 block(WAVES * 32);
  sdpa_fwd_kernel<<<grid, block, SMEM_BYTES, stream>>>(q, k, v, mask, out, attn);
}
